// DeformAttention_70772471103872
// MI455X (gfx1250) — matrix operand order repaired and checked by reading, not compiled, not run on hardware
//
#include <hip/hip_runtime.h>
#include <hip/hip_bf16.h>
#include <math.h>

// ---------------------------------------------------------------------------
// Deformable attention (DAT-style) for MI455X / gfx1250.
// wave32 + v_wmma_f32_16x16x32_bf16 + Tensor Data Mover LDS staging.
// ---------------------------------------------------------------------------

typedef __attribute__((ext_vector_type(16))) __bf16        v16bf;
typedef __attribute__((ext_vector_type(8)))  float         v8f;
typedef __attribute__((ext_vector_type(4)))  unsigned int  v4u;
typedef __attribute__((ext_vector_type(8)))  int           v8i;
typedef __attribute__((ext_vector_type(4)))  int           v4i;

#define DA_B      4
#define DA_C      512
#define DA_H      32
#define DA_W      32
#define DA_HW     1024          // H*W == nS
#define DA_HEADS  8
#define DA_HC     64
#define DA_GROUPS 4
#define DA_GC     128
#define DA_RPE    63            // 2*H-1
#define DA_SCALE  0.125f        // HC^-0.5

// GEMM tiling
#define TM 128
#define TN 128
#define TK 32

// A-matrix (16x32 bf16) per-lane K index for element e (0..15), lane-half h.
__device__ __forceinline__ int amapK(int h, int e) {
    return (e < 8) ? (h * 8 + e) : (16 + h * 8 + (e - 8));
}
// B-matrix (32x16 bf16) per-lane K index: same order as amapK.
__device__ __forceinline__ int bmapK(int h, int e) {
    return (e < 8) ? (h * 8 + e) : (16 + h * 8 + (e - 8));
}

// pack two fp32 -> packed bf16 pair (lowers to v_cvt_pk_bf16_f32)
__device__ __forceinline__ unsigned pack2bf(float a, float b) {
    union { __bf16 h; unsigned short s; } ua, ub;
    ua.h = (__bf16)a; ub.h = (__bf16)b;
    return (unsigned)ua.s | ((unsigned)ub.s << 16);
}

// ---------------------------------------------------------------------------
// Build TDM D# group0/group1 for a 2D fp32 tile (tile_d0 x tile_d1 elements,
// row stride `stride0` elements) living at `gaddr`, destined for LDS `laddr`.
// Field packing per CDNA5 ISA.
// This toolchain exposes the 6-arg builtin:
//   (uint32x4 g0, int32x8 g1, int32x4 g2, int32x4 g3, int32x8 pad, i32 cpol)
// ---------------------------------------------------------------------------
__device__ __forceinline__ void tdm_load_2d(unsigned laddr,
                                            const float* gaddr,
                                            unsigned tile_d0, unsigned tile_d1,
                                            unsigned stride0) {
    unsigned long long ga = (unsigned long long)(size_t)gaddr;
    v4u g0;
    g0[0] = 1u;                                         // count=1, user mode
    g0[1] = laddr;                                      // lds_addr
    g0[2] = (unsigned)(ga & 0xffffffffu);               // global_addr[31:0]
    g0[3] = (unsigned)((ga >> 32) & 0x01ffffffu)        // global_addr[56:32]
          | (2u << 30);                                 // type = 2 (image)
    const unsigned dim0 = 1u << 20;                     // generous OOB bounds
    const unsigned dim1 = 1u << 20;
    v8i g1;
    g1[0] = (int)(2u << 16);                            // data_size = 2 -> 4B
    g1[1] = (int)((dim0 & 0xffffu) << 16);              // tensor_dim0[15:0]
    g1[2] = (int)((dim0 >> 16) | ((dim1 & 0xffffu) << 16));
    g1[3] = (int)((dim1 >> 16) | (tile_d0 << 16));      // tile_dim0
    g1[4] = (int)(tile_d1);                             // tile_dim1 (tile_dim2=0)
    g1[5] = (int)stride0;                               // tensor_dim0_stride[31:0]
    g1[6] = 0;                                          // stride0 hi | stride1 lo
    g1[7] = 0;
    v4i z4 = {0, 0, 0, 0};
    v8i z8 = {0, 0, 0, 0, 0, 0, 0, 0};
    __builtin_amdgcn_tensor_load_to_lds(g0, g1, z4, z4, z8, 0);
}

// ---------------------------------------------------------------------------
// conv1x1 as per-batch GEMM:  Y[b,co,n] = sum_ci W[co,ci]*X[b,ci,n] + bias[co]
// 128x128 tile / 256-thread block (8 waves); TDM stages fp32 tiles into LDS,
// block converts to bf16 WMMA fragments, waves run 8 wmma per K-step.
// ---------------------------------------------------------------------------
__global__ __launch_bounds__(256)
void gemm_bf16(const float* __restrict__ X, const float* __restrict__ Wm,
               const float* __restrict__ bias, float* __restrict__ Y,
               int Cout, int Cin, int N) {
    __shared__ float sA[TM * TK];                       // fp32 A staging [128][32]
    __shared__ float sB[TK * TN];                       // fp32 B staging [32][128]
    __shared__ __align__(32) __bf16 fA[8 * 32 * 16];    // A frags [mb][lane][e]
    __shared__ __align__(32) __bf16 fB[8 * 32 * 16];    // B frags [nb][lane][e]

    const int tid  = threadIdx.x;
    const int lane = tid & 31;
    const int wave = tid >> 5;
    const int half = lane >> 4;
    const int l15  = lane & 15;
    const int m0 = blockIdx.y * TM;
    const int n0 = blockIdx.x * TN;
    const int b  = blockIdx.z;

    const float* Xb = X + (size_t)b * Cin * N;
    float*       Yb = Y + (size_t)b * Cout * N;

    const int waveM = wave >> 2;                        // 0..1 : 4 m-blocks each
    const int waveN = wave & 3;                         // 0..3 : 2 n-blocks each

    v8f acc[4][2] = {};

    const unsigned ldsA = (unsigned)(size_t)&sA[0];
    const unsigned ldsB = (unsigned)(size_t)&sB[0];

    for (int k0 = 0; k0 < Cin; k0 += TK) {
        // ---- stage A (128x32 of W) and B (32x128 of X) via Tensor Data Mover
        if (wave == 0) {
            tdm_load_2d(ldsA, Wm + (size_t)m0 * Cin + k0, TK, TM, (unsigned)Cin);
            tdm_load_2d(ldsB, Xb + (size_t)k0 * N + n0,  TN, TK, (unsigned)N);
            __builtin_amdgcn_s_wait_tensorcnt((short)0);
        }
        __syncthreads();

        // ---- fp32 staging -> bf16 WMMA fragments (packed bf16x2 stores) ----
        // A: 128 rows x 16 k-pairs = 2048 pairs
#pragma unroll
        for (int i = 0; i < 8; ++i) {
            int p  = tid + i * 256;
            int m  = p >> 4;
            int kp = (p & 15) * 2;
            float f0 = sA[m * TK + kp];
            float f1 = sA[m * TK + kp + 1];
            int L  = (m & 15) + 16 * ((kp & 15) >> 3);
            int e  = (kp & 7) | ((kp >> 4) << 3);
            int mb = m >> 4;
            *(unsigned*)&fA[((mb * 32 + L) * 16) + e] = pack2bf(f0, f1);
        }
        // B: 128 cols x 16 k-pairs = 2048 pairs
#pragma unroll
        for (int i = 0; i < 8; ++i) {
            int p  = tid + i * 256;
            int n  = p & 127;
            int kp = (p >> 7) * 2;
            float f0 = sB[kp * TN + n];
            float f1 = sB[(kp + 1) * TN + n];
            int L  = (n & 15) + 16 * ((kp & 15) >> 3);
            int e  = (kp & 7) | ((kp >> 4) << 3);
            int nb = n >> 4;
            *(unsigned*)&fB[((nb * 32 + L) * 16) + e] = pack2bf(f0, f1);
        }
        __syncthreads();

        // ---- WMMA: each wave 4 m-blocks x 2 n-blocks --------------------
        v16bf af[4];
#pragma unroll
        for (int mi = 0; mi < 4; ++mi)
            af[mi] = *(const v16bf*)&fA[((waveM * 4 + mi) * 32 + lane) * 16];
#pragma unroll
        for (int ni = 0; ni < 2; ++ni) {
            v16bf bf = *(const v16bf*)&fB[((waveN * 2 + ni) * 32 + lane) * 16];
#pragma unroll
            for (int mi = 0; mi < 4; ++mi)
                acc[mi][ni] = __builtin_amdgcn_wmma_f32_16x16x32_bf16(
                    false, af[mi], false, bf, (short)0, acc[mi][ni], false, false);
        }
        __syncthreads();                                // staging reuse next step
    }

    // ---- epilogue: C/D layout row M = v + 8*half, col N = l15 -------------
#pragma unroll
    for (int mi = 0; mi < 4; ++mi)
#pragma unroll
        for (int ni = 0; ni < 2; ++ni)
#pragma unroll
            for (int v = 0; v < 8; ++v) {
                int m = m0 + (waveM * 4 + mi) * 16 + v + 8 * half;
                int n = n0 + (waveN * 2 + ni) * 16 + l15;
                Yb[(size_t)m * N + n] = acc[mi][ni][v] + bias[m];
            }
}

// ---------------------------------------------------------------------------
// Offset path: depthwise 5x5 conv + channel LayerNorm + GELU + 1x1->2 + tanh,
// then pos = off + ref.  One 128-thread block per (bg, pixel).
// ---------------------------------------------------------------------------
__global__ __launch_bounds__(128)
void offset_kernel(const float* __restrict__ q, const float* __restrict__ dw_w,
                   const float* __restrict__ dw_b, const float* __restrict__ ln_g,
                   const float* __restrict__ ln_b, const float* __restrict__ off_w,
                   float* __restrict__ posbuf, float* __restrict__ out_pos,
                   float* __restrict__ out_ref) {
    const int c   = threadIdx.x;
    const int pix = blockIdx.x & (DA_HW - 1);
    const int bg  = blockIdx.x >> 10;
    const int b   = bg >> 2, g = bg & 3;
    const int y   = pix >> 5, x = pix & 31;

    const float* qc = q + (size_t)(b * DA_C + g * DA_GC + c) * DA_HW;

    float s = dw_b[c];
#pragma unroll
    for (int ky = 0; ky < 5; ++ky) {
        int yy = y + ky - 2;
        if (yy < 0 || yy > 31) continue;
#pragma unroll
        for (int kx = 0; kx < 5; ++kx) {
            int xx = x + kx - 2;
            if (xx < 0 || xx > 31) continue;
            s += dw_w[c * 25 + ky * 5 + kx] * qc[yy * 32 + xx];
        }
    }

    __shared__ float red[DA_GC];

    red[c] = s; __syncthreads();
    for (int o = 64; o > 0; o >>= 1) { if (c < o) red[c] += red[c + o]; __syncthreads(); }
    float mu = red[0] * (1.f / DA_GC); __syncthreads();
    red[c] = (s - mu) * (s - mu); __syncthreads();
    for (int o = 64; o > 0; o >>= 1) { if (c < o) red[c] += red[c + o]; __syncthreads(); }
    float var = red[0] * (1.f / DA_GC); __syncthreads();

    float nrm = (s - mu) * rsqrtf(var + 1e-5f) * ln_g[c] + ln_b[c];
    float gel = 0.5f * nrm * (1.f + erff(nrm * 0.70710678118654752f));

    red[c] = gel * off_w[c]; __syncthreads();
    for (int o = 64; o > 0; o >>= 1) { if (c < o) red[c] += red[c + o]; __syncthreads(); }
    float oy = red[0]; __syncthreads();
    red[c] = gel * off_w[DA_GC + c]; __syncthreads();
    for (int o = 64; o > 0; o >>= 1) { if (c < o) red[c] += red[c + o]; __syncthreads(); }
    float ox = red[0];

    if (c == 0) {
        const float rng = 2.0f / 32.0f;
        float ry = ((float)y + 0.5f) * (2.0f / DA_H) - 1.0f;
        float rx = ((float)x + 0.5f) * (2.0f / DA_W) - 1.0f;
        float py = tanhf(oy) * rng + ry;
        float px = tanhf(ox) * rng + rx;
        size_t o2 = ((size_t)bg * DA_HW + pix) * 2;
        posbuf[o2]  = py;  posbuf[o2 + 1]  = px;
        out_pos[o2] = py;  out_pos[o2 + 1] = px;
        out_ref[o2] = ry;  out_ref[o2 + 1] = rx;
    }
}

// ---------------------------------------------------------------------------
// Deformable bilinear sampling of x at pos (align_corners=True, zero pad).
// ---------------------------------------------------------------------------
__global__ __launch_bounds__(256)
void sample_kernel(const float* __restrict__ x, const float* __restrict__ posbuf,
                   float* __restrict__ xs) {
    size_t idx = (size_t)blockIdx.x * 256 + threadIdx.x;
    const size_t total = (size_t)DA_B * DA_GROUPS * DA_GC * DA_HW;
    if (idx >= total) return;
    int p  = (int)(idx & (DA_HW - 1));
    int c  = (int)((idx >> 10) & (DA_GC - 1));
    int bg = (int)(idx >> 17);
    int b = bg >> 2, g = bg & 3;

    const float* img = x + (size_t)(b * DA_C + g * DA_GC + c) * DA_HW;
    float py = posbuf[((size_t)bg * DA_HW + p) * 2];
    float px = posbuf[((size_t)bg * DA_HW + p) * 2 + 1];
    float gx = (px + 1.f) * 0.5f * (DA_W - 1);
    float gy = (py + 1.f) * 0.5f * (DA_H - 1);
    float x0f = floorf(gx), y0f = floorf(gy);
    int   x0 = (int)x0f,    y0 = (int)y0f;
    float wx = gx - x0f,    wy = gy - y0f;

    float r = 0.f;
#pragma unroll
    for (int dy = 0; dy < 2; ++dy)
#pragma unroll
        for (int dx = 0; dx < 2; ++dx) {
            int ix = x0 + dx, iy = y0 + dy;
            if (ix >= 0 && ix < DA_W && iy >= 0 && iy < DA_H) {
                float w = (dx ? wx : 1.f - wx) * (dy ? wy : 1.f - wy);
                r += img[iy * DA_W + ix] * w;
            }
        }
    xs[(size_t)(b * DA_C + g * DA_GC + c) * DA_HW + p] = r;
}

// RPE table bilinear (align_corners=True, zero pad) on a 63x63 table.
__device__ __forceinline__ float rpe_sample(const float* __restrict__ tab,
                                            float dy, float dx) {
    float gx = (dx + 1.f) * 0.5f * (DA_RPE - 1);
    float gy = (dy + 1.f) * 0.5f * (DA_RPE - 1);
    float x0f = floorf(gx), y0f = floorf(gy);
    int   x0 = (int)x0f,    y0 = (int)y0f;
    float wx = gx - x0f,    wy = gy - y0f;
    float r = 0.f;
#pragma unroll
    for (int dyi = 0; dyi < 2; ++dyi)
#pragma unroll
        for (int dxi = 0; dxi < 2; ++dxi) {
            int ix = x0 + dxi, iy = y0 + dyi;
            if (ix >= 0 && ix < DA_RPE && iy >= 0 && iy < DA_RPE) {
                float w = (dxi ? wx : 1.f - wx) * (dyi ? wy : 1.f - wy);
                r += tab[iy * DA_RPE + ix] * w;
            }
        }
    return r;
}

// ---------------------------------------------------------------------------
// Fused attention: per (head-batch, 16-row tile) wave:
//   S = (q^T k)*scale + rpe_bias  (WMMA bf16, bias via bilinear table lookup)
//   P = softmax_rows(S)           (wave32 shfl reductions, S in LDS)
//   O = P @ V^T                   (WMMA bf16, fp32 accum)
// ---------------------------------------------------------------------------
__global__ __launch_bounds__(32)
void attn_kernel(const float* __restrict__ qbuf, const float* __restrict__ kbuf,
                 const float* __restrict__ vbuf, const float* __restrict__ posbuf,
                 const float* __restrict__ rpe, float* __restrict__ obuf) {
    __shared__ float sS[16 * DA_HW];

    const int lane = threadIdx.x;
    const int half = lane >> 4;
    const int l15  = lane & 15;
    const int m0   = blockIdx.x * 16;
    const int bh   = blockIdx.y;
    const int b    = bh >> 3;
    const int head = bh & 7;
    const int g    = head >> 1;
    const int bg   = b * DA_GROUPS + g;

    const float* qh = qbuf + (size_t)(b * DA_C + head * DA_HC) * DA_HW;
    const float* kh = kbuf + (size_t)(b * DA_C + head * DA_HC) * DA_HW;
    const float* vh = vbuf + (size_t)(b * DA_C + head * DA_HC) * DA_HW;
    const float* tab  = rpe + (size_t)head * DA_RPE * DA_RPE;
    const float* posg = posbuf + (size_t)bg * DA_HW * 2;

    v16bf qa0, qa1;
    {
        const int mrow = m0 + l15;
#pragma unroll
        for (int e = 0; e < 16; ++e) {
            qa0[e] = (__bf16)qh[(size_t)(amapK(half, e)) * DA_HW + mrow];
            qa1[e] = (__bf16)qh[(size_t)(32 + amapK(half, e)) * DA_HW + mrow];
        }
    }

    for (int n0 = 0; n0 < DA_HW; n0 += 16) {
        v16bf kb0, kb1;
        const int ncol = n0 + l15;
#pragma unroll
        for (int e = 0; e < 16; ++e) {
            kb0[e] = (__bf16)kh[(size_t)(bmapK(half, e)) * DA_HW + ncol];
            kb1[e] = (__bf16)kh[(size_t)(32 + bmapK(half, e)) * DA_HW + ncol];
        }
        v8f acc = {};
        acc = __builtin_amdgcn_wmma_f32_16x16x32_bf16(
                  false, qa0, false, kb0, (short)0, acc, false, false);
        acc = __builtin_amdgcn_wmma_f32_16x16x32_bf16(
                  false, qa1, false, kb1, (short)0, acc, false, false);
#pragma unroll
        for (int v = 0; v < 8; ++v) {
            int M  = v + 8 * half;
            int gm = m0 + M;
            int gn = n0 + l15;
            float ry = ((float)(gm >> 5) + 0.5f) * (2.0f / DA_H) - 1.0f;
            float rx = ((float)(gm & 31) + 0.5f) * (2.0f / DA_W) - 1.0f;
            float dy = (ry - posg[gn * 2])     * 0.5f;
            float dx = (rx - posg[gn * 2 + 1]) * 0.5f;
            float bias = rpe_sample(tab, dy, dx);
            sS[M * DA_HW + gn] = acc[v] * DA_SCALE + bias;
        }
    }
    __syncthreads();

    for (int r = 0; r < 16; ++r) {
        float mx = -1e30f;
        for (int ccol = lane; ccol < DA_HW; ccol += 32)
            mx = fmaxf(mx, sS[r * DA_HW + ccol]);
#pragma unroll
        for (int o = 16; o > 0; o >>= 1) mx = fmaxf(mx, __shfl_xor(mx, o, 32));
        float sum = 0.f;
        for (int ccol = lane; ccol < DA_HW; ccol += 32) {
            float e = __expf(sS[r * DA_HW + ccol] - mx);
            sS[r * DA_HW + ccol] = e;
            sum += e;
        }
#pragma unroll
        for (int o = 16; o > 0; o >>= 1) sum += __shfl_xor(sum, o, 32);
        float inv = 1.f / sum;
        for (int ccol = lane; ccol < DA_HW; ccol += 32)
            sS[r * DA_HW + ccol] *= inv;
    }
    __syncthreads();

#pragma unroll
    for (int t = 0; t < 4; ++t) {
        v8f oacc = {};
        for (int nb = 0; nb < DA_HW; nb += 32) {
            v16bf pa, vb;
#pragma unroll
            for (int e = 0; e < 16; ++e) {
                pa[e] = (__bf16)sS[l15 * DA_HW + nb + amapK(half, e)];
                vb[e] = (__bf16)vh[(size_t)(t * 16 + l15) * DA_HW + nb + bmapK(half, e)];
            }
            oacc = __builtin_amdgcn_wmma_f32_16x16x32_bf16(
                       false, pa, false, vb, (short)0, oacc, false, false);
        }
#pragma unroll
        for (int v = 0; v < 8; ++v) {
            int m = m0 + v + 8 * half;
            obuf[(size_t)(b * DA_C + head * DA_HC + t * 16 + l15) * DA_HW + m] = oacc[v];
        }
    }
}

// ---------------------------------------------------------------------------
// Host launcher
// ---------------------------------------------------------------------------
extern "C" void kernel_launch(void* const* d_in, const int* in_sizes, int n_in,
                              void* d_out, int out_size, void* d_ws, size_t ws_size,
                              hipStream_t stream) {
    const float* x    = (const float*)d_in[0];
    const float* w_q  = (const float*)d_in[1];
    const float* b_q  = (const float*)d_in[2];
    const float* w_k  = (const float*)d_in[3];
    const float* b_k  = (const float*)d_in[4];
    const float* w_v  = (const float*)d_in[5];
    const float* b_v  = (const float*)d_in[6];
    const float* w_o  = (const float*)d_in[7];
    const float* b_o  = (const float*)d_in[8];
    const float* dw_w = (const float*)d_in[9];
    const float* dw_b = (const float*)d_in[10];
    const float* ln_g = (const float*)d_in[11];
    const float* ln_b = (const float*)d_in[12];
    const float* off_w = (const float*)d_in[13];
    const float* rpe  = (const float*)d_in[14];

    float* out = (float*)d_out;
    const size_t Y_ELEMS   = (size_t)DA_B * DA_C * DA_HW;
    const size_t POS_ELEMS = (size_t)DA_B * DA_GROUPS * DA_HW * 2;
    float* out_y   = out;
    float* out_pos = out + Y_ELEMS;
    float* out_ref = out + Y_ELEMS + POS_ELEMS;

    float* ws = (float*)d_ws;
    float* qbuf   = ws;
    float* posbuf = qbuf   + Y_ELEMS;
    float* xsbuf  = posbuf + POS_ELEMS;
    float* kbuf   = xsbuf  + Y_ELEMS;
    float* vbuf   = kbuf   + Y_ELEMS;
    float* obuf   = vbuf   + Y_ELEMS;

    dim3 gemmGrid(DA_HW / TN, DA_C / TM, DA_B);   // (8, 4, 4)

    gemm_bf16<<<gemmGrid, 256, 0, stream>>>(x, w_q, b_q, qbuf, DA_C, DA_C, DA_HW);

    offset_kernel<<<DA_B * DA_GROUPS * DA_HW, DA_GC, 0, stream>>>(
        qbuf, dw_w, dw_b, ln_g, ln_b, off_w, posbuf, out_pos, out_ref);

    {
        size_t total = (size_t)DA_B * DA_GROUPS * DA_GC * DA_HW;
        sample_kernel<<<(unsigned)((total + 255) / 256), 256, 0, stream>>>(
            x, posbuf, xsbuf);
    }

    gemm_bf16<<<gemmGrid, 256, 0, stream>>>(xsbuf, w_k, b_k, kbuf, DA_C, DA_C, DA_HW);
    gemm_bf16<<<gemmGrid, 256, 0, stream>>>(xsbuf, w_v, b_v, vbuf, DA_C, DA_C, DA_HW);

    attn_kernel<<<dim3(DA_HW / 16, DA_B * DA_HEADS), 32, 0, stream>>>(
        qbuf, kbuf, vbuf, posbuf, rpe, obuf);

    gemm_bf16<<<gemmGrid, 256, 0, stream>>>(obuf, w_o, b_o, out_y, DA_C, DA_C, DA_HW);
}
